// SelfAttention_60129542633
// MI455X (gfx1250) — matrix operand order repaired and checked by reading, not compiled, not run on hardware
//
#include <hip/hip_runtime.h>
#include <hip/hip_bf16.h>

typedef __attribute__((ext_vector_type(16))) _Float16     v16h;
typedef __attribute__((ext_vector_type(8)))  float        v8f;
typedef __attribute__((ext_vector_type(4)))  int          i4;
typedef __attribute__((ext_vector_type(4)))  unsigned int u32x4;
typedef __attribute__((ext_vector_type(8)))  int          i32x8;
typedef __attribute__((ext_vector_type(4)))  int          i32x4;

#define BATCH 4
#define CH    256
#define MIDC  32
#define NPOS  4096   // 64*64
#define JBLK  128    // j positions staged per iteration

#if __has_builtin(__builtin_amdgcn_tensor_load_to_lds) && \
    __has_builtin(__builtin_amdgcn_s_wait_tensorcnt)
#define USE_TDM 1
#else
#define USE_TDM 0
#endif
#if !USE_TDM && __has_builtin(__builtin_amdgcn_global_load_async_to_lds_b128) && \
    __has_builtin(__builtin_amdgcn_s_wait_asynccnt)
#define USE_ASYNC_LDS 1
#else
#define USE_ASYNC_LDS 0
#endif

union F16Frag { v16h h; unsigned int u[8]; };

// ---------------------------------------------------------------------------
// Kernel 1: fused QKV 1x1-conv GEMM. One wave per 16x16 tile; row-tile index
// in blockIdx.y so weight/bias/store selection is wave-uniform (SALU branch).
// Rows [0,256) -> v (C,N); [256,288) -> q (N,mid); [288,320) -> k (N,mid).
// ---------------------------------------------------------------------------
__global__ void qkv_kernel(const float* __restrict__ x,
                           const float* __restrict__ Wq, const float* __restrict__ bq,
                           const float* __restrict__ Wk, const float* __restrict__ bk,
                           const float* __restrict__ Wv, const float* __restrict__ bv,
                           _Float16* __restrict__ qo, _Float16* __restrict__ ko,
                           _Float16* __restrict__ vo) {
    const int lane = threadIdx.x;            // 0..31 (wave32)
    const int rt   = blockIdx.y;             // 0..19 row-tiles (uniform)
    const int n0   = blockIdx.x * 16;
    const int b    = blockIdx.z;
    const int iL   = lane & 15;
    const int hf   = lane >> 4;

    const float* xb = x + (size_t)b * CH * NPOS;

    const float* Wrow; const float* bias; int rbase; int kind;
    if (rt < 16)      { Wrow = Wv + (size_t)rt * 16 * CH;        bias = bv; rbase = rt * 16;        kind = 0; }
    else if (rt < 18) { Wrow = Wq + (size_t)(rt - 16) * 16 * CH; bias = bq; rbase = (rt - 16) * 16; kind = 1; }
    else              { Wrow = Wk + (size_t)(rt - 18) * 16 * CH; bias = bk; rbase = (rt - 18) * 16; kind = 2; }

    v8f acc = {};
    for (int cc = 0; cc < CH; cc += 32) {
        v16h a, bm;
#pragma unroll
        for (int vv = 0; vv < 8; ++vv) {
            // A-fragment (16x32 f16): row = lane&15, K = (v>>2)*16 + hf*8 + (v&3)*2
            const int Ka = ((vv >> 2) << 4) + (hf << 3) + ((vv & 3) << 1);
            const float* wp = Wrow + (size_t)iL * CH + cc + Ka;
            a[2 * vv]     = (_Float16)wp[0];
            a[2 * vv + 1] = (_Float16)wp[1];
            // B-fragment (32x16 f16): col = lane&15, K = (v>>2)*16 + hf*8 + (v&3)*2
            const int Kb = ((vv >> 2) << 4) + (hf << 3) + ((vv & 3) << 1);
            bm[2 * vv]     = (_Float16)xb[(size_t)(cc + Kb) * NPOS + n0 + iL];
            bm[2 * vv + 1] = (_Float16)xb[(size_t)(cc + Kb + 1) * NPOS + n0 + iL];
        }
        acc = __builtin_amdgcn_wmma_f32_16x16x32_f16(false, a, false, bm,
                                                     (short)0, acc, false, false);
    }

    const int col = n0 + iL;
    if (kind == 0) {
#pragma unroll
        for (int r = 0; r < 8; ++r) {
            const int rowm = rbase + r + (hf << 3);
            vo[((size_t)b * CH + rowm) * NPOS + col] = (_Float16)(acc[r] + bias[rowm]);
        }
    } else if (kind == 1) {
#pragma unroll
        for (int r = 0; r < 8; ++r) {
            const int rowm = rbase + r + (hf << 3);
            qo[((size_t)b * NPOS + col) * MIDC + rowm] = (_Float16)(acc[r] + bias[rowm]);
        }
    } else {
#pragma unroll
        for (int r = 0; r < 8; ++r) {
            const int rowm = rbase + r + (hf << 3);
            ko[((size_t)b * NPOS + col) * MIDC + rowm] = (_Float16)(acc[r] + bias[rowm]);
        }
    }
}

// ---------------------------------------------------------------------------
// Kernel 2: fused streaming attention + residual.
// One WG (8 waves) per 16-query tile per batch. Per 128-j block:
//   produce: each wave -> one 16x16 S^T tile (WMMA), exp, P^T -> LDS + rowsum
//   consume: each wave -> O(32ch x 16i) += V * P^T; V chunks double-buffered
//            into LDS via TDM tensor_load_to_lds (TENSORcnt) when available,
//            else GLOBAL_LOAD_ASYNC_TO_LDS_B128 (ASYNCcnt), else direct loads.
// Finalize: out = x + gamma * O / l.
// ---------------------------------------------------------------------------
__global__ void attn_kernel(const _Float16* __restrict__ qb_,
                            const _Float16* __restrict__ kb_,
                            const _Float16* __restrict__ vb_,
                            const float* __restrict__ x,
                            const float* __restrict__ gamma,
                            float* __restrict__ out) {
    __shared__ _Float16 Ps[16][JBLK + 4];   // P^T staged as [i][j], j contiguous
    __shared__ float lsum[16];              // softmax denominators per query i
#if USE_TDM || USE_ASYNC_LDS
    __shared__ _Float16 Vs[2][8][32][32];   // per-wave double-buffered V chunks
#endif

    const int tid  = threadIdx.x;
    const int lane = tid & 31;
    const int wave = tid >> 5;              // 0..7
    const int iL   = lane & 15;
    const int hf   = lane >> 4;
    const int b    = blockIdx.y;
    const int i0   = blockIdx.x * 16;
    const int c0   = wave * 32;

    const _Float16* qp = qb_ + (size_t)b * NPOS * MIDC;
    const _Float16* kp = kb_ + (size_t)b * NPOS * MIDC;
    const _Float16* vp = vb_ + (size_t)b * CH * NPOS;

    if (tid < 16) lsum[tid] = 0.0f;

    // loop-invariant Q B-fragment: col = query i, K = m (pairs contiguous)
    F16Frag qf;
    {
        const _Float16* qrow = qp + (size_t)(i0 + iL) * MIDC;
#pragma unroll
        for (int vv = 0; vv < 8; ++vv)
            qf.u[vv] = *(const unsigned int*)(qrow + ((vv >> 2) << 4) + (hf << 3) + ((vv & 3) << 1));
    }

#if USE_TDM
    // TDM: one 2D descriptor DMAs a 32ch x 32j f16 tile (row stride N) to LDS.
    // D# group1 is fully constant: data_size=2B, tensor=(N x C), tile=32x32,
    // dim0_stride=N. Group0 carries lds/global addresses (wave-uniform).
    const i32x8 g1c = { (int)(1u << 16),                    // data_size=1 (2B)
                        (int)(((unsigned)NPOS & 0xFFFFu) << 16), // tensor_dim0 lo16
                        (int)((unsigned)CH << 16),          // dim0 hi16=0 | tensor_dim1 lo16
                        (int)(32u << 16),                   // dim1 hi16=0 | tile_dim0=32
                        32,                                 // tile_dim1=32, tile_dim2=0
                        NPOS,                               // tensor_dim0_stride lo32
                        0, 0 };                             // stride0 hi, stride1
    const i32x4 gz  = { 0, 0, 0, 0 };
    const i32x8 gz8 = { 0, 0, 0, 0, 0, 0, 0, 0 };
    auto issueV = [&](int buf, int jcol0) {
        unsigned long long ga = (unsigned long long)(const void*)
            (vp + (size_t)c0 * NPOS + jcol0);
        unsigned lds_off = (unsigned)(unsigned long long)
            (__attribute__((address_space(3))) _Float16*)&Vs[buf][wave][0][0];
        u32x4 g0;
        g0.x = 1u;                                          // count=1, user mode
        g0.y = (unsigned)__builtin_amdgcn_readfirstlane((int)lds_off);
        g0.z = (unsigned)__builtin_amdgcn_readfirstlane((int)(unsigned)ga);
        g0.w = (unsigned)__builtin_amdgcn_readfirstlane(
                   (int)(((unsigned)(ga >> 32) & 0x1FFFFFFu) | (2u << 30)));
        __builtin_amdgcn_tensor_load_to_lds(g0, g1c, gz, gz, gz8, 0);
    };
#elif USE_ASYNC_LDS
    // async-copy one 32ch x 32j f16 V chunk: one row per lane, 4 x B128
    auto issueV = [&](int buf, int jcol0) {
        const _Float16* src = vp + (size_t)(c0 + lane) * NPOS + jcol0;
#pragma unroll
        for (int part = 0; part < 4; ++part)
            __builtin_amdgcn_global_load_async_to_lds_b128(
                (__attribute__((address_space(1))) i4*)(src + part * 8),
                (__attribute__((address_space(3))) i4*)&Vs[buf][wave][lane][part * 8],
                0, 0);
    };
#endif

    v8f o0 = {}, o1 = {};
    const float scale = 0.17677669529663687f;   // 1/sqrt(mid)
    __syncthreads();

    for (int jb = 0; jb < NPOS; jb += JBLK) {
#if USE_TDM || USE_ASYNC_LDS
        issueV(0, jb);                          // overlap with S production
#endif
        // ---- produce P^T for 16 j owned by this wave ----
        const int j0 = jb + wave * 16;
        F16Frag ka;
        const _Float16* krow = kp + (size_t)(j0 + iL) * MIDC;
#pragma unroll
        for (int vv = 0; vv < 8; ++vv) {
            const int Ka = ((vv >> 2) << 4) + (hf << 3) + ((vv & 3) << 1);
            ka.u[vv] = *(const unsigned int*)(krow + Ka);
        }
        if (jb + JBLK < NPOS)   // prefetch next K block -> global_prefetch_b8
            __builtin_prefetch(kp + (size_t)(jb + JBLK + wave * 16 + iL) * MIDC, 0, 1);

        v8f s = {};
        s = __builtin_amdgcn_wmma_f32_16x16x32_f16(false, ka.h, false, qf.h,
                                                   (short)0, s, false, false);
        float part = 0.0f;
#pragma unroll
        for (int r = 0; r < 8; ++r) {
            const float e = __expf(s[r] * scale);     // S^T[j,i], j = r + hf*8
            part += e;
            Ps[iL][wave * 16 + r + (hf << 3)] = (_Float16)e;
        }
        atomicAdd(&lsum[iL], part);                   // ds_add_f32
        __syncthreads();

        // ---- consume: O(32ch x 16i) += V(ch x 32j) * P^T(32j x 16i) ----
#pragma unroll
        for (int chn = 0; chn < 4; ++chn) {
            F16Frag pb, va0, va1;
#if USE_TDM || USE_ASYNC_LDS
            if (chn < 3) issueV((chn + 1) & 1, jb + (chn + 1) * 32);
#if USE_TDM
            if (chn < 3) __builtin_amdgcn_s_wait_tensorcnt(1);  // chunk chn landed
            else         __builtin_amdgcn_s_wait_tensorcnt(0);
#else
            if (chn < 3) __builtin_amdgcn_s_wait_asynccnt(4);   // chunk chn landed
            else         __builtin_amdgcn_s_wait_asynccnt(0);
#endif
            const int bufc = chn & 1;
#pragma unroll
            for (int vv = 0; vv < 8; ++vv) {
                pb.u[vv] = *(const unsigned int*)&Ps[iL][chn * 32 + ((vv >> 2) << 4) + (hf << 3) + ((vv & 3) << 1)];
                const int Ka = ((vv >> 2) << 4) + (hf << 3) + ((vv & 3) << 1);
                va0.u[vv] = *(const unsigned int*)&Vs[bufc][wave][iL][Ka];
                va1.u[vv] = *(const unsigned int*)&Vs[bufc][wave][16 + iL][Ka];
            }
#else
#pragma unroll
            for (int vv = 0; vv < 8; ++vv) {
                pb.u[vv] = *(const unsigned int*)&Ps[iL][chn * 32 + ((vv >> 2) << 4) + (hf << 3) + ((vv & 3) << 1)];
                const int Ka = ((vv >> 2) << 4) + (hf << 3) + ((vv & 3) << 1);
                const _Float16* vrow = vp + (size_t)(c0 + iL) * NPOS + jb + chn * 32 + Ka;
                va0.u[vv] = *(const unsigned int*)vrow;
                va1.u[vv] = *(const unsigned int*)(vrow + (size_t)16 * NPOS);
            }
#endif
            o0 = __builtin_amdgcn_wmma_f32_16x16x32_f16(false, va0.h, false, pb.h,
                                                        (short)0, o0, false, false);
            o1 = __builtin_amdgcn_wmma_f32_16x16x32_f16(false, va1.h, false, pb.h,
                                                        (short)0, o1, false, false);
        }
        __syncthreads();
    }

    // ---- finalize: out = x + gamma * O / l ----
    const float inv = 1.0f / lsum[iL];
    const float g   = gamma[0];
    const float* xb = x   + (size_t)b * CH * NPOS;
    float*       ob = out + (size_t)b * CH * NPOS;
    const int col = i0 + iL;
#pragma unroll
    for (int r = 0; r < 8; ++r) {
        const int c = c0 + r + (hf << 3);
        ob[(size_t)c * NPOS + col]        = xb[(size_t)c * NPOS + col]        + g * o0[r] * inv;
        ob[(size_t)(c + 16) * NPOS + col] = xb[(size_t)(c + 16) * NPOS + col] + g * o1[r] * inv;
    }
}

// ---------------------------------------------------------------------------
extern "C" void kernel_launch(void* const* d_in, const int* in_sizes, int n_in,
                              void* d_out, int out_size, void* d_ws, size_t ws_size,
                              hipStream_t stream) {
    const float* x     = (const float*)d_in[0];
    const float* Wq    = (const float*)d_in[1];
    const float* bq    = (const float*)d_in[2];
    const float* Wk    = (const float*)d_in[3];
    const float* bk    = (const float*)d_in[4];
    const float* Wv    = (const float*)d_in[5];
    const float* bv    = (const float*)d_in[6];
    const float* gamma = (const float*)d_in[7];

    // workspace: q (B,N,32) f16 | k (B,N,32) f16 | v (B,C,N) f16  (~10 MB)
    _Float16* qw = (_Float16*)d_ws;
    _Float16* kw = qw + (size_t)BATCH * NPOS * MIDC;
    _Float16* vw = kw + (size_t)BATCH * NPOS * MIDC;

    qkv_kernel<<<dim3(NPOS / 16, 20, BATCH), dim3(32), 0, stream>>>(
        x, Wq, bq, Wk, bk, Wv, bv, qw, kw, vw);

    attn_kernel<<<dim3(NPOS / 16, BATCH), 256, 0, stream>>>(
        qw, kw, vw, x, gamma, (float*)d_out);
}
